// attention_layer_89867895702135
// MI455X (gfx1250) — matrix operand order repaired and checked by reading, not compiled, not run on hardware
//
#include <hip/hip_runtime.h>
#include <stdint.h>

#define BB 8
#define NPTS 32768
#define NF 256
#define PER_B (NPTS * NF)   // floats per batch

typedef __attribute__((ext_vector_type(16))) __bf16 v16bf;
typedef __attribute__((ext_vector_type(2)))  __bf16 v2bf;
typedef __attribute__((ext_vector_type(8)))  float  v8f;

struct U32x8 { uint4 a, b; };

static __device__ __forceinline__ v16bf as_v16bf(const U32x8 &s) {
  v16bf v; __builtin_memcpy(&v, &s, 32); return v;
}
static __device__ __forceinline__ v16bf load_pk16(const unsigned short *p) {
  U32x8 t = *(const U32x8 *)p; return as_v16bf(t);
}
// f32 pair -> packed bf16 dword (RNE).  Prefer the dedicated builtin; else use
// compiler fptrunc casts so the backend can select a native v_cvt_*bf16*
// instruction if the target has one (falls back to the RNE integer sequence
// only if the ISA truly lacks it).
static __device__ __forceinline__ unsigned pk2bf(float a, float b) {
#if __has_builtin(__builtin_amdgcn_cvt_pk_bf16_f32)
  v2bf v = __builtin_amdgcn_cvt_pk_bf16_f32(a, b);
  unsigned u; __builtin_memcpy(&u, &v, 4);
  return u;
#else
  v2bf v;
  v[0] = (__bf16)a;
  v[1] = (__bf16)b;
  unsigned u; __builtin_memcpy(&u, &v, 4);
  return u;
#endif
}
static __device__ __forceinline__ v8f zero8() {
  v8f z = {0.f, 0.f, 0.f, 0.f, 0.f, 0.f, 0.f, 0.f}; return z;
}
static __device__ __forceinline__ v8f wmma_bf16(v16bf a, v16bf b, v8f c) {
  return __builtin_amdgcn_wmma_f32_16x16x32_bf16(false, a, false, b, (short)0, c,
                                                 false, false);
}

// K0: pack Wq|Wk (32x256) and Wv (256x256) into bf16 WMMA A-tile layout.
// A-tile per-lane: m = lane&15, hi = lane>>4; k(h)=h+16*hi
__global__ __launch_bounds__(32) void k0_pack_w(const float *Wq, const float *Wk,
                                                const float *Wv,
                                                unsigned short *wqkPk,
                                                unsigned short *wvPk) {
  int id = blockIdx.x, lane = threadIdx.x;
  int m = lane & 15, hi = lane >> 4;
  const float *W;
  unsigned short *dst;
  if (id < 16) {
    int mt = id >> 3, kt = id & 7;
    int row = mt * 16 + m;
    W = (row < 16) ? (Wq + row * NF) : (Wk + (row - 16) * NF);
    W += kt * 32 + 16 * hi;
    dst = wqkPk + (id * 512 + lane * 16);
  } else {
    int tid = id - 16;
    int mt = tid >> 3, kt = tid & 7;
    W = Wv + (mt * 16 + m) * NF + kt * 32 + 16 * hi;
    dst = wvPk + (tid * 512 + lane * 16);
  }
  float f[16];
#pragma unroll
  for (int j = 0; j < 8; j++) { f[j] = W[j]; f[8 + j] = W[8 + j]; }
  unsigned u[8];
#pragma unroll
  for (int j = 0; j < 8; j++) u[j] = pk2bf(f[2 * j], f[2 * j + 1]);
  U32x8 t; __builtin_memcpy(&t, u, 32);
  ((uint4 *)dst)[0] = t.a;
  ((uint4 *)dst)[1] = t.b;
}

// K1: [Q;K](32x32768) = Wqk(32x256) @ X(256x32768) per batch (bf16 WMMA),
// biases folded in, f32 results to ws.
__global__ __launch_bounds__(256) void k1_qk(const float *x, const float *bq,
                                             const float *bk,
                                             const unsigned short *wqkPk,
                                             float *qBuf, float *kBuf) {
  int b = blockIdx.x >> 7, pblk = blockIdx.x & 127;
  int p0 = pblk * 256;
  int w = threadIdx.x >> 5, lane = threadIdx.x & 31;
  int nn = lane & 15, hi = lane >> 4;
  const float *xb = x + (size_t)b * PER_B;

#pragma unroll
  for (int ni = 0; ni < 2; ni++) {
    int ntl = 2 * w + ni;
    int p = p0 + ntl * 16 + nn;
    v16bf Bt[8];
#pragma unroll
    for (int kt = 0; kt < 8; kt++) {
      int c0 = kt * 32 + 16 * hi;        // B layout: k = h + 16*hi
      float f[16];
#pragma unroll
      for (int j = 0; j < 16; j++) f[j] = xb[(size_t)(c0 + j) * NPTS + p];
      unsigned u[8];
#pragma unroll
      for (int j = 0; j < 8; j++) u[j] = pk2bf(f[2 * j], f[2 * j + 1]);
      U32x8 t; __builtin_memcpy(&t, u, 32);
      Bt[kt] = as_v16bf(t);
    }
#pragma unroll
    for (int mti = 0; mti < 2; mti++) {
      v8f C = zero8();
#pragma unroll
      for (int kt = 0; kt < 8; kt++) {
        v16bf A = load_pk16(wqkPk + ((mti * 8 + kt) * 512 + lane * 16));
        C = wmma_bf16(A, Bt[kt], C);
      }
      float *dst = (mti == 0) ? qBuf : kBuf;
      const float *bias = (mti == 0) ? bq : bk;
#pragma unroll
      for (int v = 0; v < 8; v++) {
        int f0 = v + 8 * hi;             // C layout row
        dst[((size_t)b * 16 + f0) * NPTS + p] = C[v] + bias[f0];
      }
    }
  }
}

// K2: energy[b,f,g] = sum_p2 Q[f,p2]*K[p2>>11,(p2&2047)*16+g]  (raw .view of k)
__global__ __launch_bounds__(256) void k2_zero(float *e) {
  e[blockIdx.x * 256 + threadIdx.x] = 0.f;
}

__global__ __launch_bounds__(256) void k2_energy(const float *qBuf,
                                                 const float *kBuf,
                                                 float *eBuf) {
  int b = blockIdx.x >> 5, chunk = blockIdx.x & 31;
  int f = threadIdx.x >> 4, g = threadIdx.x & 15;
  const float *Q = qBuf + ((size_t)b * 16 + f) * NPTS;
  const float *K = kBuf + (size_t)b * 16 * NPTS;
  float acc = 0.f;
  int p2_0 = chunk * 1024;
#pragma unroll 4
  for (int i = 0; i < 1024; i++) {
    int p2 = p2_0 + i;
    acc += Q[p2] * K[(size_t)(p2 >> 11) * NPTS + ((p2 & 2047) * 16) + g];
  }
  atomicAdd(&eBuf[(b * 16 + f) * 16 + g], acc);
}

// K3: closed-form stride-16 ConvTranspose upsample + column softmax.
// Writes attention f32 to d_out, plus bf16 B-tile-packed copy to ws.
// B-tile: element (kk,nn) at lane = nn + 16*(kk>>4), h = kk&15.
__global__ __launch_bounds__(256) void k3_att(const float *eBuf, const float *Wup,
                                              const float *bup_p, float *attOut,
                                              unsigned short *attPk) {
  __shared__ float eL[256];
  __shared__ float wL[9];
  int b = blockIdx.x;
  int g = threadIdx.x;
  eL[g] = eBuf[b * 256 + g];
  if (g < 9) wL[g] = Wup[g];
  __syncthreads();

  float bb = bup_p[0];
  int j = g >> 4, s = g & 15;
  float ea[48];
  float ebase = 0.f, inv_d = 0.f;
  if (s < 3) {
    float a[48], mx = bb;
#pragma unroll
    for (int i = 0; i < 16; i++)
#pragma unroll
      for (int r = 0; r < 3; r++) {
        float v = eL[i * 16 + j] * wL[r * 3 + s] + bb;
        a[i * 3 + r] = v;
        mx = fmaxf(mx, v);
      }
    ebase = __expf(bb - mx);
    float d = 208.f * ebase;              // 13 plain rows per 16-block * 16
#pragma unroll
    for (int t = 0; t < 48; t++) { ea[t] = __expf(a[t] - mx); d += ea[t]; }
    inv_d = 1.f / d;
  }

  float *outCol = attOut + (size_t)b * 65536;
  int nt = g >> 4, nn = g & 15;
#pragma unroll
  for (int kt = 0; kt < 8; kt++) {
    float vals[32];
#pragma unroll
    for (int kk = 0; kk < 32; kk++) {
      int fch = kt * 32 + kk;
      int i = fch >> 4, r = fch & 15;
      float av = (s < 3) ? (((r < 3) ? ea[i * 3 + r] : ebase) * inv_d)
                         : (1.f / 256.f);
      vals[kk] = av;
      outCol[fch * 256 + g] = av;
    }
    unsigned short *base = attPk + (size_t)((b * 8 + kt) * 16 + nt) * 512;
    unsigned u[8];
    U32x8 t;
#pragma unroll
    for (int j2 = 0; j2 < 8; j2++) u[j2] = pk2bf(vals[2 * j2], vals[2 * j2 + 1]);
    __builtin_memcpy(&t, u, 32);
    ((uint4 *)(base + nn * 16))[0] = t.a;
    ((uint4 *)(base + nn * 16))[1] = t.b;
#pragma unroll
    for (int j2 = 0; j2 < 8; j2++)
      u[j2] = pk2bf(vals[16 + 2 * j2], vals[17 + 2 * j2]);
    __builtin_memcpy(&t, u, 32);
    ((uint4 *)(base + (nn + 16) * 16))[0] = t.a;
    ((uint4 *)(base + (nn + 16) * 16))[1] = t.b;
  }
}

// K4: fused  Out_slice(b,pb) = gamma*(Wv @ (Xb @ att) + bv) + Xb
// 512 threads = 16 waves; wave w owns output row-tile w. 256KB dynamic LDS.
__global__ __launch_bounds__(512) void k4_main(const float *x, const float *bv,
                                               const float *gamma_p,
                                               const unsigned short *wvPk,
                                               const unsigned short *attPk,
                                               float *out) {
  extern __shared__ unsigned short smem[];   // 256 KB
  unsigned short *attLds = smem;             // 65536 bf16 (packed B tiles)
  unsigned short *Mp = smem + 65536;         // 65536 bf16 (packed B tiles)

  int b = blockIdx.x >> 7, pb = blockIdx.x & 127;
  int w = threadIdx.x >> 5;                  // wave id == mtile
  int lane = threadIdx.x & 31;
  int nn = lane & 15, lh = lane >> 4;
  const float *xb = x + (size_t)b * PER_B;
  float gamma = gamma_p[0];

  // stage this batch's packed attention into LDS.
  // Prefer the CDNA5 async global->LDS DMA path (no VGPR round trip,
  // tracked by ASYNCcnt, overlaps with the A-tile conversion below).
  {
#if __has_builtin(__builtin_amdgcn_global_load_async_to_lds_b128)
    typedef int v4i_t __attribute__((ext_vector_type(4)));
    typedef __attribute__((address_space(1))) v4i_t *gv4i_p;
    typedef __attribute__((address_space(3))) v4i_t *lv4i_p;
    char *src = (char *)(attPk + (size_t)b * 65536);
    char *dst = (char *)attLds;
#pragma unroll
    for (int it = 0; it < 16; it++) {
      int i = threadIdx.x + it * 512;      // 8192 x 16B = 128 KB
      __builtin_amdgcn_global_load_async_to_lds_b128(
          (gv4i_p)(src + (size_t)i * 16), (lv4i_p)(dst + (size_t)i * 16), 0, 0);
    }
#else
    const uint4 *srcv = (const uint4 *)(attPk + (size_t)b * 65536);
    uint4 *dls = (uint4 *)attLds;
    for (int i = threadIdx.x; i < 8192; i += 512) dls[i] = srcv[i];
#endif
  }

  // GEMM1 A tiles: Xb rows c = w*16 + (lane&15), converted f32->bf16
  // (overlaps with the async att staging above)
  v16bf A[8];
  {
    const float *src = xb + (size_t)(w * 16 + nn) * NPTS + (size_t)pb * 256;
#pragma unroll
    for (int kt = 0; kt < 8; kt++) {
      int c0 = kt * 32 + 16 * lh;            // A layout k-runs
      float4 f0 = *(const float4 *)(src + c0);
      float4 f1 = *(const float4 *)(src + c0 + 4);
      float4 f2 = *(const float4 *)(src + c0 + 8);
      float4 f3 = *(const float4 *)(src + c0 + 12);
      unsigned u[8];
      u[0] = pk2bf(f0.x, f0.y); u[1] = pk2bf(f0.z, f0.w);
      u[2] = pk2bf(f1.x, f1.y); u[3] = pk2bf(f1.z, f1.w);
      u[4] = pk2bf(f2.x, f2.y); u[5] = pk2bf(f2.z, f2.w);
      u[6] = pk2bf(f3.x, f3.y); u[7] = pk2bf(f3.z, f3.w);
      U32x8 t; __builtin_memcpy(&t, u, 32);
      A[kt] = as_v16bf(t);
    }
  }
#if __has_builtin(__builtin_amdgcn_global_load_async_to_lds_b128)
#if __has_builtin(__builtin_amdgcn_s_wait_asynccnt)
  __builtin_amdgcn_s_wait_asynccnt(0);
#else
  asm volatile("s_wait_asynccnt 0" ::: "memory");
#endif
#endif
  __syncthreads();

  // GEMM1: M row-tile w = Xb @ att
#pragma unroll 1
  for (int nt = 0; nt < 16; nt++) {
    v8f C = zero8();
#pragma unroll
    for (int kt = 0; kt < 8; kt++) {
      v16bf Bt = load_pk16(attLds + ((kt * 16 + nt) * 512 + lane * 16));
      C = wmma_bf16(A[kt], Bt, C);
    }
    // C tile -> Mp in B layout: k = w*16+v+8*lh -> lane' = nn+16*(w&1), h=v+8*lh
    unsigned u[4];
    u[0] = pk2bf(C[0], C[1]); u[1] = pk2bf(C[2], C[3]);
    u[2] = pk2bf(C[4], C[5]); u[3] = pk2bf(C[6], C[7]);
    uint4 q; __builtin_memcpy(&q, u, 16);
    int idx = ((w >> 1) * 16 + nt) * 512 + (nn + 16 * (w & 1)) * 16 + 8 * lh;
    *(uint4 *)(Mp + idx) = q;
  }
  __syncthreads();

  // GEMM2: Out row-tile w = Wv @ M, fused bias/residual/gamma epilogue
#pragma unroll
  for (int kt = 0; kt < 8; kt++)
    A[kt] = load_pk16(wvPk + ((w * 8 + kt) * 512 + lane * 16));
  float bvv[8];
#pragma unroll
  for (int v = 0; v < 8; v++) bvv[v] = bv[w * 16 + v + 8 * lh];

#pragma unroll 1
  for (int nt = 0; nt < 16; nt++) {
    v8f C = zero8();
#pragma unroll
    for (int kt = 0; kt < 8; kt++) {
      v16bf Bt = load_pk16(Mp + ((kt * 16 + nt) * 512 + lane * 16));
      C = wmma_bf16(A[kt], Bt, C);
    }
    int g = nt * 16 + nn;
#pragma unroll
    for (int v = 0; v < 8; v++) {
      int o = w * 16 + v + 8 * lh;
      size_t off = (size_t)b * PER_B + (size_t)o * NPTS + (size_t)pb * 256 + g;
      out[off] = gamma * (C[v] + bvv[v]) + x[off];
    }
  }
}

extern "C" void kernel_launch(void *const *d_in, const int *in_sizes, int n_in,
                              void *d_out, int out_size, void *d_ws,
                              size_t ws_size, hipStream_t stream) {
  (void)in_sizes; (void)n_in; (void)out_size; (void)ws_size;
  const float *x     = (const float *)d_in[0];
  const float *Wq    = (const float *)d_in[1];
  const float *bq    = (const float *)d_in[2];
  const float *Wk    = (const float *)d_in[3];
  const float *bk    = (const float *)d_in[4];
  const float *Wv    = (const float *)d_in[5];
  const float *bv    = (const float *)d_in[6];
  const float *Wup   = (const float *)d_in[7];
  const float *bup   = (const float *)d_in[8];
  const float *gamma = (const float *)d_in[9];

  float *out = (float *)d_out;
  float *attOut = out + (size_t)BB * PER_B;   // attention appended after out

  char *ws = (char *)d_ws;
  float *qBuf = (float *)ws;                                           // 16 MB
  float *kBuf = (float *)(ws + (size_t)16 * 1024 * 1024);              // 16 MB
  float *eBuf = (float *)(ws + (size_t)32 * 1024 * 1024);              // 8 KB
  unsigned short *attPk =
      (unsigned short *)(ws + (size_t)32 * 1024 * 1024 + 64 * 1024);   // 1 MB
  unsigned short *wvPk =
      (unsigned short *)(ws + (size_t)32 * 1024 * 1024 + 64 * 1024 +
                         1024 * 1024);                                  // 128 KB
  unsigned short *wqkPk =
      (unsigned short *)(ws + (size_t)32 * 1024 * 1024 + 64 * 1024 +
                         1024 * 1024 + 128 * 1024);                     // 16 KB

  k0_pack_w<<<144, 32, 0, stream>>>(Wq, Wk, Wv, wqkPk, wvPk);
  k1_qk<<<BB * 128, 256, 0, stream>>>(x, bq, bk, wqkPk, qBuf, kBuf);
  k2_zero<<<8, 256, 0, stream>>>(eBuf);
  k2_energy<<<BB * 32, 256, 0, stream>>>(qBuf, kBuf, eBuf);
  k3_att<<<BB, 256, 0, stream>>>(eBuf, Wup, bup, attOut, attPk);
  k4_main<<<BB * 128, 512, 262144, stream>>>(x, bv, gamma, wvPk, attPk, out);
}
